// EncoderRNN_72172630442564
// MI455X (gfx1250) — matrix operand order repaired and checked by reading, not compiled, not run on hardware
//
#include <hip/hip_runtime.h>
#include <hip/hip_bf16.h>

#define B_TOT 65536
#define T_SEQ 10
#define HDIM  128
#define HSTRIDE 144   // h-scratch row stride in bf16 elems (288B: 16B-aligned, breaks bank conflicts)

typedef __attribute__((ext_vector_type(16))) __bf16 bf16x16;
typedef __attribute__((ext_vector_type(8)))  __bf16 bf16x8;
typedef __attribute__((ext_vector_type(8)))  float  f32x8;
typedef __attribute__((ext_vector_type(4)))  float  f32x4;

static_assert(sizeof(bf16x16) == 32, "bf16x16 must be 32B");

__device__ __forceinline__ float sigf(float x){ return 1.0f/(1.0f + __expf(-x)); }
__device__ __forceinline__ float tanhx(float x){ return 1.0f - 2.0f/(1.0f + __expf(2.0f*x)); }

__device__ __forceinline__ bf16x16 catbf(bf16x8 lo, bf16x8 hi){
  return __builtin_shufflevector(lo, hi, 0,1,2,3,4,5,6,7,8,9,10,11,12,13,14,15);
}

__device__ __forceinline__ void wait_async0(){
#if __has_builtin(__builtin_amdgcn_s_wait_asynccnt)
  __builtin_amdgcn_s_wait_asynccnt(0);
#else
  asm volatile("s_wait_asynccnt 0" ::: "memory");
#endif
}

// Async DMA copy global -> LDS; 16B per lane per issue, tracked by ASYNCcnt.
__device__ __forceinline__ void stage_bytes(const unsigned char* __restrict__ src,
                                            unsigned char* dst_lds, int nbytes, int tid){
  for (int off = tid*16; off < nbytes; off += 256*16){
    unsigned int lds_off = (unsigned int)(unsigned long long)(dst_lds + off);
    asm volatile("global_load_async_to_lds_b128 %0, %1, off"
                 :: "v"(lds_off), "v"(src + off) : "memory");
  }
}

// ---------------------------------------------------------------------------
// Weight prep: bf16 B-fragment blobs, layout [jc(8)][gt(4)][kk(KT)][lane(32)][j(16)]:
//   N = gt*128 + jc*16 + (lane%16),  K = kk*32 + (lane/16)*8 + (j%8) + (j/8)*16
// Layer0 Kcat = 32 (padded emb) + 128 (h) = 160 -> KT=5.
// Layer1 Kcat = 128 (y0) + 128 (h) = 256       -> KT=8.
// ---------------------------------------------------------------------------
template<int LAYER, int KT>
__global__ void prep_w(const float* __restrict__ Wih, const float* __restrict__ Whh,
                       __bf16* __restrict__ blob){
  int i = blockIdx.x*blockDim.x + threadIdx.x;
  const int TOT = 8*4*KT*32*16;
  if (i >= TOT) return;
  int j    =  i        & 15;
  int lane = (i >> 4)  & 31;
  int kk   = (i >> 9)  % KT;
  int gt   = (i / (512*KT)) & 3;
  int jc   =  i / (2048*KT);
  int n = gt*128 + jc*16 + (lane & 15);
  int k = kk*32 + (lane >> 4)*8 + (j & 7) + (j >> 3)*16;
  float v;
  if (LAYER == 0) {
    if (k < 32) v = (k < 30) ? Wih[n*30 + k] : 0.0f;
    else        v = Whh[n*128 + (k - 32)];
  } else {
    if (k < 128) v = Wih[n*128 + k];
    else         v = Whh[n*128 + (k - 128)];
  }
  blob[i] = (__bf16)v;
}

__global__ void prep_bias(const float* __restrict__ bi0, const float* __restrict__ bh0,
                          const float* __restrict__ bi1, const float* __restrict__ bh1,
                          float* __restrict__ bias0, float* __restrict__ bias1){
  int i = blockIdx.x*blockDim.x + threadIdx.x;
  if (i < 512)       bias0[i]       = bi0[i]       + bh0[i];
  else if (i < 1024) bias1[i - 512] = bi1[i - 512] + bh1[i - 512];
}

// ---------------------------------------------------------------------------
// One LSTM layer over all T steps, weights fully LDS-resident.
// Block = 256 threads = 8 waves; wave w owns batch rows [blk*128+16w, +16).
// Startup: one async DMA burst stages the whole bf16 weight blob + bias into
// LDS, one barrier. The time loop then has NO barriers: h-state is
// wave-private (LDS scratch for D-layout -> A-layout transpose), c stays in
// registers, gates via v_wmma_f32_16x16x32_bf16 with f32 accumulation.
// ---------------------------------------------------------------------------
template<int LAYER>
__global__ __launch_bounds__(256, 1)
void lstm_pass(const unsigned char* __restrict__ wblob,
               const float* __restrict__ bias,
               const float* __restrict__ h0,
               const float* __restrict__ c0,
               float* __restrict__ yio,
               float* __restrict__ hn,
               float* __restrict__ cn,
               const int* __restrict__ car_idx,
               const int* __restrict__ region_idx,
               const int* __restrict__ poi_idx,
               const float* __restrict__ car_emb,
               const float* __restrict__ region_emb,
               const float* __restrict__ poi_emb)
{
  constexpr int KT     = (LAYER == 0) ? 5 : 8;     // K tiles of 32 in [x|h]
  constexpr int XT     = (LAYER == 0) ? 1 : 4;     // K tiles belonging to x
  constexpr int WBYTES = 8*4*KT*1024;              // full blob: 160KB / 256KB

  __shared__ __align__(32) unsigned char wlds[WBYTES];
  __shared__ __align__(32) __bf16 hbuf[8][16*HSTRIDE];   // ~36KB, wave-private
  __shared__ __align__(16) float  blds[512];             // combined bias

  const int tid  = threadIdx.x;
  const int w    = tid >> 5;
  const int lane = tid & 31;
  const int seg  = lane >> 4;        // 0/1 half of wave
  const int lm   = lane & 15;
  const int rowBase = blockIdx.x*128 + w*16;
  const int myRow   = rowBase + lm;  // A-frag row for this lane
  const int k0      = seg*8;         // A-frag K sub-offset

  // Kick off the weight + bias DMA immediately.
  stage_bytes(wblob, wlds, WBYTES, tid);
  stage_bytes((const unsigned char*)bias, (unsigned char*)blds, 2048, tid);

  // While DMA is in flight: c into regs (D-layout), h0 into LDS scratch (bf16).
  float creg[8][8];
  #pragma unroll
  for (int jc = 0; jc < 8; ++jc) {
    #pragma unroll
    for (int r = 0; r < 8; ++r) {
      int rrow = r + 8*seg;
      int row  = rowBase + rrow;
      int col  = jc*16 + lm;
      creg[jc][r] = c0[row*HDIM + col];
      hbuf[w][rrow*HSTRIDE + col] = (__bf16)h0[row*HDIM + col];
    }
  }

  wait_async0();
  __syncthreads();   // whole blob + bias now visible to all waves

  for (int t = 0; t < T_SEQ; ++t) {
    // ---- preload all A-fragments for this step (regs, reused over jc,gt) ----
    bf16x16 afr[KT];
    if (LAYER == 0) {
      float v[16];
      int ib = myRow*T_SEQ + t;
      #pragma unroll
      for (int s = 0; s < 2; ++s)
        #pragma unroll
        for (int j = 0; j < 8; ++j) {
          int k = k0 + s*16 + j;
          float x = 0.0f;
          if (k < 10)      x = car_emb   [car_idx   [ib]*10 + k];
          else if (k < 20) x = region_emb[region_idx[ib]*10 + (k-10)];
          else if (k < 30) x = poi_emb   [poi_idx   [ib]*10 + (k-20)];
          v[s*8 + j] = x;
        }
      bf16x16 a;
      #pragma unroll
      for (int e = 0; e < 16; ++e) a[e] = (__bf16)v[e];
      afr[0] = a;
    } else {
      const float* xp = yio + ((size_t)myRow*T_SEQ + t)*HDIM;
      #pragma unroll
      for (int kk = 0; kk < XT; ++kk) {
        f32x4 q0 = *(const f32x4*)(xp + kk*32 + k0);
        f32x4 q1 = *(const f32x4*)(xp + kk*32 + k0 + 4);
        f32x4 q2 = *(const f32x4*)(xp + kk*32 + k0 + 16);
        f32x4 q3 = *(const f32x4*)(xp + kk*32 + k0 + 20);
        bf16x16 a;
        #pragma unroll
        for (int j = 0; j < 4; ++j) {
          a[j]      = (__bf16)q0[j];
          a[4 + j]  = (__bf16)q1[j];
          a[8 + j]  = (__bf16)q2[j];
          a[12 + j] = (__bf16)q3[j];
        }
        afr[kk] = a;
      }
    }
    #pragma unroll
    for (int kk = XT; kk < KT; ++kk) {
      int kh = (kk - XT)*32 + k0;
      const __bf16* hp = &hbuf[w][lm*HSTRIDE + kh];
      bf16x8 lo = *(const bf16x8*)hp;
      bf16x8 hi = *(const bf16x8*)(hp + 16);
      afr[kk] = catbf(lo, hi);
    }

    // ---- gates for each hidden chunk jc (cols jc*16..+15 of each gate) ----
    #pragma unroll
    for (int jc = 0; jc < 8; ++jc) {
      f32x8 acc[4];
      #pragma unroll
      for (int g = 0; g < 4; ++g) {
        float bv = blds[g*128 + jc*16 + lm];
        f32x8 a = { bv, bv, bv, bv, bv, bv, bv, bv };
        acc[g] = a;
      }
      #pragma unroll
      for (int kk = 0; kk < KT; ++kk) {
        #pragma unroll
        for (int gt = 0; gt < 4; ++gt) {
          const __bf16* wp =
              (const __bf16*)(wlds + (((jc*4 + gt)*KT + kk) << 10) + lane*32);
          bf16x8 lo = *(const bf16x8*)wp;
          bf16x8 hi = *(const bf16x8*)(wp + 8);
          bf16x16 bf = catbf(lo, hi);
          acc[gt] = __builtin_amdgcn_wmma_f32_16x16x32_bf16(
              false, afr[kk], false, bf, (short)0, acc[gt], false, false);
        }
      }
      // ---- elementwise LSTM update for hidden cols [jc*16, jc*16+16) ----
      #pragma unroll
      for (int r = 0; r < 8; ++r) {
        float iv = sigf (acc[0][r]);
        float fv = sigf (acc[1][r]);
        float gv = tanhx(acc[2][r]);
        float ov = sigf (acc[3][r]);
        float cnew = fv*creg[jc][r] + iv*gv;
        creg[jc][r] = cnew;
        float hnew = ov*tanhx(cnew);
        int rrow = r + 8*seg;
        int row  = rowBase + rrow;
        int col  = jc*16 + lm;
        hbuf[w][rrow*HSTRIDE + col] = (__bf16)hnew;        // next-step A source
        yio[((size_t)row*T_SEQ + t)*HDIM + col] = hnew;    // layer output
        if (t == T_SEQ - 1) {
          hn[(size_t)row*HDIM + col] = hnew;
          cn[(size_t)row*HDIM + col] = cnew;
        }
      }
    }
  }
}

// ---------------------------------------------------------------------------
extern "C" void kernel_launch(void* const* d_in, const int* in_sizes, int n_in,
                              void* d_out, int out_size, void* d_ws, size_t ws_size,
                              hipStream_t stream) {
  (void)in_sizes; (void)n_in; (void)out_size; (void)ws_size;

  const int*   car_idx    = (const int*)  d_in[0];
  const int*   region_idx = (const int*)  d_in[1];
  const int*   poi_idx    = (const int*)  d_in[2];
  const float* h0         = (const float*)d_in[3];   // [2,B,H]
  const float* c0         = (const float*)d_in[4];   // [2,B,H]
  const float* car_emb    = (const float*)d_in[5];
  const float* region_emb = (const float*)d_in[6];
  const float* poi_emb    = (const float*)d_in[7];
  const float* W_ih0      = (const float*)d_in[8];   // [512,30]
  const float* W_hh0      = (const float*)d_in[9];   // [512,128]
  const float* b_ih0      = (const float*)d_in[10];
  const float* b_hh0      = (const float*)d_in[11];
  const float* W_ih1      = (const float*)d_in[12];  // [512,128]
  const float* W_hh1      = (const float*)d_in[13];  // [512,128]
  const float* b_ih1      = (const float*)d_in[14];
  const float* b_hh1      = (const float*)d_in[15];

  float* y  = (float*)d_out;                          // [B,T,H]
  float* hn = y  + (size_t)B_TOT*T_SEQ*HDIM;          // [2,B,H]
  float* cn = hn + 2*(size_t)B_TOT*HDIM;              // [2,B,H]

  // workspace: bf16 weight blobs + combined biases (~420 KB)
  unsigned char* ws = (unsigned char*)d_ws;
  __bf16* blob0 = (__bf16*)ws;                        // 8*4*5*512 = 81920 elems
  __bf16* blob1 = (__bf16*)(ws + 163840);             // 8*4*8*512 = 131072 elems
  float*  bias0 = (float*)(ws + 163840 + 262144);
  float*  bias1 = bias0 + 512;

  prep_w<0,5><<<(81920  + 255)/256, 256, 0, stream>>>(W_ih0, W_hh0, blob0);
  prep_w<1,8><<<(131072 + 255)/256, 256, 0, stream>>>(W_ih1, W_hh1, blob1);
  prep_bias<<<4, 256, 0, stream>>>(b_ih0, b_hh0, b_ih1, b_hh1, bias0, bias1);

  const size_t BH = (size_t)B_TOT*HDIM;
  // Layer 0: x = embeddings; writes y0 into the y region (scratch reuse).
  lstm_pass<0><<<B_TOT/128, 256, 0, stream>>>(
      (const unsigned char*)blob0, bias0, h0, c0, y, hn, cn,
      car_idx, region_idx, poi_idx, car_emb, region_emb, poi_emb);
  // Layer 1: reads y0 from y region, overwrites with y1 in place.
  lstm_pass<1><<<B_TOT/128, 256, 0, stream>>>(
      (const unsigned char*)blob1, bias1, h0 + BH, c0 + BH, y, hn + BH, cn + BH,
      nullptr, nullptr, nullptr, nullptr, nullptr, nullptr);
}
